// GConvNet_71588514889986
// MI455X (gfx1250) — matrix operand order repaired and checked by reading, not compiled, not run on hardware
//
#include <hip/hip_runtime.h>
#include <hip/hip_bf16.h>

// ---------------- problem constants ----------------
#define NN    20000          // nodes
#define EE    120000         // edges
#define TMAXC (1 << 20)      // triplet capacity (E^2/N ~ 720K expected)
#define DMAXC 65536          // degree histogram size
#define HALFN (NN / 2)

typedef __attribute__((ext_vector_type(8)))  float  v8f;
typedef __attribute__((ext_vector_type(16))) __bf16 v16bf;

struct U32x8 { uint4 lo, hi; };   // 32B carrier for a v16bf fragment

__device__ __forceinline__ unsigned short f2bf(float f) {
    unsigned u = __builtin_bit_cast(unsigned, f);
    unsigned r = u + 0x7FFFu + ((u >> 16) & 1u);   // round-to-nearest-even
    return (unsigned short)(r >> 16);
}

// ---------------- generic helpers ----------------
__global__ void zero_k(int* p, long n) {
    long i = (long)blockIdx.x * blockDim.x + threadIdx.x;
    long stride = (long)gridDim.x * blockDim.x;
    for (; i < n; i += stride) p[i] = 0;
}

// single-block exclusive scan, out[n] = total (used for small n and pass 2)
__global__ void scan_k(const int* in, int* out, int n) {
    __shared__ int buf[1024];
    __shared__ int carry;
    if (threadIdx.x == 0) carry = 0;
    __syncthreads();
    for (int base = 0; base < n; base += 1024) {
        int i = base + threadIdx.x;
        int v = (i < n) ? in[i] : 0;
        buf[threadIdx.x] = v;
        __syncthreads();
        for (int off = 1; off < 1024; off <<= 1) {
            int t = (threadIdx.x >= off) ? buf[threadIdx.x - off] : 0;
            __syncthreads();
            buf[threadIdx.x] += t;
            __syncthreads();
        }
        if (i < n) out[i] = buf[threadIdx.x] - v + carry;
        __syncthreads();
        if (threadIdx.x == 0) carry += buf[1023];
        __syncthreads();
    }
    if (threadIdx.x == 0) out[n] = carry;
}

// hierarchical scan pass 1: per-block local exclusive scan + block totals
__global__ void scanA_k(const int* __restrict__ in, int* __restrict__ out,
                        int* __restrict__ bsum, int n) {
    __shared__ int buf[1024];
    int i = blockIdx.x * 1024 + threadIdx.x;
    int v = (i < n) ? in[i] : 0;
    buf[threadIdx.x] = v;
    __syncthreads();
    for (int off = 1; off < 1024; off <<= 1) {
        int t = (threadIdx.x >= off) ? buf[threadIdx.x - off] : 0;
        __syncthreads();
        buf[threadIdx.x] += t;
        __syncthreads();
    }
    if (i < n) out[i] = buf[threadIdx.x] - v;
    if (threadIdx.x == 1023) bsum[blockIdx.x] = buf[1023];
}
// hierarchical scan pass 3: add block offsets, publish total at out[n]
__global__ void scanC_k(int* __restrict__ out, const int* __restrict__ boff, int n) {
    int i = blockIdx.x * 1024 + threadIdx.x;
    if (i < n) out[i] += boff[blockIdx.x];
    if (i == 0) out[n] = boff[gridDim.x];
}

// ---------------- index pipeline ----------------
__global__ void hist_tgt_k(const int* ei, int* degT) {
    int e = blockIdx.x * blockDim.x + threadIdx.x;
    for (; e < EE; e += gridDim.x * blockDim.x)
        atomicAdd(degT + ei[EE + e], 1);
}
__global__ void fill_bytgt_k(const int* ei, const int* offT, int* curT, int* byTgt) {
    int e = blockIdx.x * blockDim.x + threadIdx.x;
    for (; e < EE; e += gridDim.x * blockDim.x) {
        int t = ei[EE + e];
        byTgt[offT[t] + atomicAdd(curT + t, 1)] = e;
    }
}
__global__ void cntE_k(const int* ei, const int* degT, int* cntE) {
    int e = blockIdx.x * blockDim.x + threadIdx.x;
    for (; e < EE; e += gridDim.x * blockDim.x)
        cntE[e] = degT[ei[e]];
}
__global__ void emit_tri_k(const int* ei, const int* degT, const int* offT,
                           const int* byTgt, const int* offE,
                           int* t0o, int* t1o, int* t2o) {
    int e = blockIdx.x * blockDim.x + threadIdx.x;
    for (; e < EE; e += gridDim.x * blockDim.x) {
        int s = ei[e], t = ei[EE + e];
        int d = degT[s], b0 = offE[e], p = offT[s];
        for (int j = 0; j < d; ++j) {
            int idx = b0 + j;
            if (idx >= TMAXC) break;
            int h = byTgt[p + j];
            t0o[idx] = ei[h]; t1o[idx] = s; t2o[idx] = t;
        }
    }
}
__global__ void hist0_k(const int* tri0, const int* offE, int* h0) {
    int T = offE[EE]; if (T > TMAXC) T = TMAXC;
    int i = blockIdx.x * blockDim.x + threadIdx.x;
    for (; i < T; i += gridDim.x * blockDim.x)
        atomicAdd(h0 + tri0[i], 1);
}
__global__ void scatter_sort_k(const int* t0i, const int* t1i, const int* t2i,
                               const int* start0, int* cur0,
                               int* s0, int* s1, int* s2, const int* offE) {
    int T = offE[EE]; if (T > TMAXC) T = TMAXC;
    int i = blockIdx.x * blockDim.x + threadIdx.x;
    for (; i < T; i += gridDim.x * blockDim.x) {
        int v = t0i[i];
        int p = start0[v] + atomicAdd(cur0 + v, 1);
        s0[p] = v; s1[p] = t1i[i]; s2[p] = t2i[i];
    }
}
__global__ void deg_hist_k(const int* s0, const int* s1, const int* s2,
                           const int* offE, int* deg) {
    int T = offE[EE]; if (T > TMAXC) T = TMAXC;
    int i = blockIdx.x * blockDim.x + threadIdx.x;
    for (; i < T; i += gridDim.x * blockDim.x) {
        atomicAdd(deg + s0[i], 1);
        atomicAdd(deg + s1[i], 1);
        atomicAdd(deg + s2[i], 1);
    }
}
__global__ void node_deg_hist_k(const int* deg, int* Dh) {
    int v = blockIdx.x * blockDim.x + threadIdx.x;
    if (v < NN) {
        int d = deg[v]; if (d > DMAXC - 1) d = DMAXC - 1;
        atomicAdd(Dh + d, 1);
    }
}
__global__ void find_thresh_k(const int* cumD, int* meta) {
    if (blockIdx.x == 0 && threadIdx.x == 0) {
        for (int d = 0; d < DMAXC; ++d) {
            if (cumD[d] <= HALFN && cumD[d + 1] > HALFN) {
                meta[1] = d;                    // d*
                meta[2] = HALFN - cumD[d];      // R ties to drop
                break;
            }
        }
    }
}
__global__ void tieflag_k(const int* deg, const int* meta, int* tf) {
    int v = blockIdx.x * blockDim.x + threadIdx.x;
    if (v < NN) tf[v] = (deg[v] == meta[1]) ? 1 : 0;
}
__global__ void keep_k(const int* deg, const int* meta, const int* tiescan, int* keep) {
    int v = blockIdx.x * blockDim.x + threadIdx.x;
    if (v < NN) {
        int ds = meta[1], R = meta[2];
        keep[v] = (deg[v] > ds) || (deg[v] == ds && tiescan[v] >= R);
    }
}
__global__ void mask_k(const int* s0, const int* s1, const int* s2,
                       const int* keep, int* mask, const int* offE) {
    int T = offE[EE]; if (T > TMAXC) T = TMAXC;
    int i = blockIdx.x * blockDim.x + threadIdx.x;
    for (; i < T; i += gridDim.x * blockDim.x)
        mask[i] = (keep[s0[i]] && keep[s1[i]] && keep[s2[i]]) ? 1 : 0;
}
__global__ void compact_k(const int* s0, const int* s1, const int* s2,
                          const int* mask, const int* mscan,
                          int* k0, int* k1, int* k2, const int* offE) {
    int T = offE[EE]; if (T > TMAXC) T = TMAXC;
    int i = blockIdx.x * blockDim.x + threadIdx.x;
    for (; i < T; i += gridDim.x * blockDim.x)
        if (mask[i]) { int p = mscan[i]; k0[p] = s0[i]; k1[p] = s1[i]; k2[p] = s2[i]; }
}
__global__ void segflag_k(const int* k0, const int* mscan, int* flag) {
    int Tk = mscan[TMAXC];
    int i = blockIdx.x * blockDim.x + threadIdx.x;
    for (; i < Tk; i += gridDim.x * blockDim.x)
        flag[i] = (i == 0 || k0[i] != k0[i - 1]) ? 1 : 0;
}
__global__ void inv_cnt_k(const int* flag, const int* segscan, int* inv,
                          float* cnt, const int* mscan) {
    int Tk = mscan[TMAXC];
    int i = blockIdx.x * blockDim.x + threadIdx.x;
    for (; i < Tk; i += gridDim.x * blockDim.x) {
        int iv = segscan[i] + flag[i] - 1;
        inv[i] = iv;
        atomicAdd(cnt + iv, 1.0f);
    }
}
__global__ void present_k(const int* k0, const int* k1, const int* k2,
                          const int* mscan, int* present) {
    int Tk = mscan[TMAXC];
    int i = blockIdx.x * blockDim.x + threadIdx.x;
    for (; i < Tk; i += gridDim.x * blockDim.x) {
        present[k0[i]] = 1; present[k1[i]] = 1; present[k2[i]] = 1;
    }
}

// ---------------- bf16 prep ----------------
__global__ void xb_prep_k(const float* x, unsigned short* xb) {
    int i = blockIdx.x * blockDim.x + threadIdx.x;
    if (i < NN * 64) xb[i] = f2bf(x[i]);
}
// Bswz[((kk*4+nt)*32 + lane)*16 + j] = W[o, i, k] with K = kk*32 + (lane>>4)*8 + (j<8 ? j : 8+j)
__global__ void w_prep_k(const float* W, unsigned short* Bswz) {
    int tid = blockIdx.x * blockDim.x + threadIdx.x;
    if (tid < 768) {
        int lane = tid & 31, fi = tid >> 5;
        int nt = fi & 3, kk = fi >> 2;
        int o = nt * 16 + (lane & 15), hi = lane >> 4;
        for (int j = 0; j < 16; ++j) {
            int K = kk * 32 + hi * 8 + (j < 8 ? j : 8 + j);
            int k = K >> 6, i = K & 63;
            Bswz[tid * 16 + j] = f2bf(W[(o * 64 + i) * 3 + k]);
        }
    }
}

// ---------------- WMMA GEMM + segment accumulate ----------------
// Per wave-tile: 16 triplets x 64 outputs, K=192 -> 6 x 4 = 24 v_wmma_f32_16x16x32_bf16
__global__ __launch_bounds__(256) void gemm_k(const unsigned short* __restrict__ xb,
                                              const uint4* __restrict__ gB,
                                              const int* __restrict__ k0,
                                              const int* __restrict__ k1,
                                              const int* __restrict__ k2,
                                              const int* __restrict__ inv,
                                              const int* __restrict__ mscan,
                                              float* __restrict__ S) {
    __shared__ uint4 ldsB[1536];   // 24 KB pre-swizzled B fragments
    for (int i = threadIdx.x; i < 1536; i += 256) ldsB[i] = gB[i];
    __syncthreads();

    const int Tk = mscan[TMAXC];
    const int wave = threadIdx.x >> 5;
    const int lane = threadIdx.x & 31;
    const int m = lane & 15, hi = lane >> 4;
    const int col = lane & 15;

    for (int tile = blockIdx.x * 8 + wave; tile * 16 < Tk; tile += gridDim.x * 8) {
        const int t0 = tile * 16;
        int t = t0 + m;
        int n0 = 0, n1 = 0, n2 = 0;
        if (t < Tk) { n0 = k0[t]; n1 = k1[t]; n2 = k2[t]; }

        v8f acc[4] = {v8f{}, v8f{}, v8f{}, v8f{}};

#pragma unroll
        for (int kk = 0; kk < 6; ++kk) {
            // A fragment: two 16B runs per lane, runs never cross a node row
            int K0 = kk * 32 + hi * 8;
            int K1 = K0 + 16;
            int c0 = K0 >> 6, c1 = K1 >> 6;
            int off0 = ((c0 == 0) ? n0 : (c0 == 1) ? n1 : n2) * 64 + (K0 & 63);
            int off1 = ((c1 == 0) ? n0 : (c1 == 1) ? n1 : n2) * 64 + (K1 & 63);
            U32x8 au;
            au.lo = *(const uint4*)(xb + off0);
            au.hi = *(const uint4*)(xb + off1);
            v16bf a = __builtin_bit_cast(v16bf, au);

#pragma unroll
            for (int nt = 0; nt < 4; ++nt) {
                int fi = ((kk * 4 + nt) * 32 + lane) * 2;
                U32x8 bu;
                bu.lo = ldsB[fi];
                bu.hi = ldsB[fi + 1];
                v16bf bfrag = __builtin_bit_cast(v16bf, bu);
                acc[nt] = __builtin_amdgcn_wmma_f32_16x16x32_bf16(
                    false, a, false, bfrag, (short)0, acc[nt], false, false);
            }
        }

        // D layout: lane column = lane&15; VGPR v -> row v + hi*8
        int segs[8];
#pragma unroll
        for (int v = 0; v < 8; ++v) {
            int tr = t0 + v + hi * 8;
            segs[v] = (tr < Tk) ? inv[tr] : -1;
        }
#pragma unroll
        for (int nt = 0; nt < 4; ++nt) {
            int o = nt * 16 + col;
#pragma unroll
            for (int v = 0; v < 8; ++v)
                if (segs[v] >= 0) atomicAdd(S + segs[v] * 64 + o, acc[nt][v]);
        }
    }
}

// ---------------- finalize ----------------
__global__ void xnew_k(const float* S, const float* cnt, const float* b,
                       const int* segscan, float* out) {
    int U = segscan[TMAXC];
    int idx = blockIdx.x * blockDim.x + threadIdx.x;
    if (idx < NN * 64) {
        int u = idx >> 6, o = idx & 63;
        if (u < U) out[idx] = S[idx] / cnt[u] + b[o];
    }
}
__global__ void newei_k(const int* k0, const int* k1, const int* k2,
                        const int* newscan, const int* mscan,
                        const int* segscan, int* outI) {
    int Tk = mscan[TMAXC];
    int U = segscan[TMAXC];
    int i = blockIdx.x * blockDim.x + threadIdx.x;
    for (; i < Tk; i += gridDim.x * blockDim.x) {
        outI[U * 64 + 0 * Tk + i] = newscan[k0[i]];
        outI[U * 64 + 1 * Tk + i] = newscan[k1[i]];
        outI[U * 64 + 2 * Tk + i] = newscan[k2[i]];
    }
}

// ---------------- host driver ----------------
extern "C" void kernel_launch(void* const* d_in, const int* in_sizes, int n_in,
                              void* d_out, int out_size, void* d_ws, size_t ws_size,
                              hipStream_t stream) {
    const float* x  = (const float*)d_in[0];
    const int*   ei = (const int*)d_in[1];
    const float* W  = (const float*)d_in[2];
    const float* b  = (const float*)d_in[3];
    float* out = (float*)d_out;
    (void)in_sizes; (void)n_in; (void)out_size; (void)ws_size;

    char* ws = (char*)d_ws;
    size_t off = 0;
    auto alloc = [&](size_t bytes) -> char* {
        char* p = ws + off;
        off = (off + bytes + 255) & ~(size_t)255;
        return p;
    };
    // zeroed region
    int* degT    = (int*)alloc(NN * 4);
    int* offT    = (int*)alloc((NN + 1) * 4);
    int* curT    = (int*)alloc(NN * 4);
    int* byTgt   = (int*)alloc(EE * 4);
    int* cntE    = (int*)alloc(EE * 4);
    int* offE    = (int*)alloc((EE + 1) * 4);
    int* tri0    = (int*)alloc((size_t)TMAXC * 4);
    int* tri1    = (int*)alloc((size_t)TMAXC * 4);
    int* tri2    = (int*)alloc((size_t)TMAXC * 4);
    int* h0      = (int*)alloc(NN * 4);
    int* start0  = (int*)alloc((NN + 1) * 4);
    int* cur0    = (int*)alloc(NN * 4);
    int* s0      = (int*)alloc((size_t)TMAXC * 4);
    int* s1      = (int*)alloc((size_t)TMAXC * 4);
    int* s2      = (int*)alloc((size_t)TMAXC * 4);
    int* deg     = (int*)alloc(NN * 4);
    int* Dh      = (int*)alloc((size_t)DMAXC * 4);
    int* cumD    = (int*)alloc((size_t)(DMAXC + 1) * 4);
    int* meta    = (int*)alloc(64);
    int* tieflag = (int*)alloc(NN * 4);
    int* tiescan = (int*)alloc((NN + 1) * 4);
    int* keep    = (int*)alloc(NN * 4);
    int* maskA   = (int*)alloc((size_t)TMAXC * 4);
    int* mscan   = (int*)alloc(((size_t)TMAXC + 1) * 4);
    int* kk0     = (int*)alloc((size_t)TMAXC * 4);
    int* kk1     = (int*)alloc((size_t)TMAXC * 4);
    int* kk2     = (int*)alloc((size_t)TMAXC * 4);
    int* segflag = (int*)alloc((size_t)TMAXC * 4);
    int* segscan = (int*)alloc(((size_t)TMAXC + 1) * 4);
    int* invA    = (int*)alloc((size_t)TMAXC * 4);
    int* present = (int*)alloc(NN * 4);
    int* newscan = (int*)alloc((NN + 1) * 4);
    float* cnt   = (float*)alloc(NN * 4);
    float* S     = (float*)alloc((size_t)NN * 64 * 4);
    size_t zeroBytes = off;
    // overwritten-every-call region (no zeroing required)
    unsigned short* xb   = (unsigned short*)alloc((size_t)NN * 64 * 2);
    unsigned short* Bswz = (unsigned short*)alloc(12288 * 2);
    int* bsum = (int*)alloc(1024 * 4);       // hierarchical-scan scratch
    int* boff = (int*)alloc(1025 * 4);

    const int B = 256;
    const int gE   = (EE + B - 1) / B;
    const int gN   = (NN + B - 1) / B;
    const int gN64 = (NN * 64 + B - 1) / B;
    const int gT   = 4096;   // grid-stride over TMAXC

    // hierarchical exclusive scan: out[0..n], out[n] = total
    auto scan3 = [&](const int* in, int* outp, int n) {
        int nb = (n + 1023) / 1024;
        scanA_k<<<nb, 1024, 0, stream>>>(in, outp, bsum, n);
        scan_k<<<1, 1024, 0, stream>>>(bsum, boff, nb);
        scanC_k<<<nb, 1024, 0, stream>>>(outp, boff, n);
    };

    zero_k<<<4096, B, 0, stream>>>((int*)ws, (long)(zeroBytes / 4));

    // CSR by tgt, triplet counts/offsets
    hist_tgt_k<<<gE, B, 0, stream>>>(ei, degT);
    scan3(degT, offT, NN);
    fill_bytgt_k<<<gE, B, 0, stream>>>(ei, offT, curT, byTgt);
    cntE_k<<<gE, B, 0, stream>>>(ei, degT, cntE);
    scan3(cntE, offE, EE);
    // emit triplets
    emit_tri_k<<<gE, B, 0, stream>>>(ei, degT, offT, byTgt, offE, tri0, tri1, tri2);
    // counting sort by tri0
    hist0_k<<<gT, B, 0, stream>>>(tri0, offE, h0);
    scan3(h0, start0, NN);
    scatter_sort_k<<<gT, B, 0, stream>>>(tri0, tri1, tri2, start0, cur0, s0, s1, s2, offE);
    // degree prune (median cut with stable tie-break)
    deg_hist_k<<<gT, B, 0, stream>>>(s0, s1, s2, offE, deg);
    node_deg_hist_k<<<gN, B, 0, stream>>>(deg, Dh);
    scan3(Dh, cumD, DMAXC);
    find_thresh_k<<<1, 1, 0, stream>>>(cumD, meta);
    tieflag_k<<<gN, B, 0, stream>>>(deg, meta, tieflag);
    scan3(tieflag, tiescan, NN);
    keep_k<<<gN, B, 0, stream>>>(deg, meta, tiescan, keep);
    // mask + compact
    mask_k<<<gT, B, 0, stream>>>(s0, s1, s2, keep, maskA, offE);
    scan3(maskA, mscan, TMAXC);
    compact_k<<<gT, B, 0, stream>>>(s0, s1, s2, maskA, mscan, kk0, kk1, kk2, offE);
    // segments (unique of kk0) + counts
    segflag_k<<<gT, B, 0, stream>>>(kk0, mscan, segflag);
    scan3(segflag, segscan, TMAXC);
    inv_cnt_k<<<gT, B, 0, stream>>>(segflag, segscan, invA, cnt, mscan);
    // reindex map
    present_k<<<gT, B, 0, stream>>>(kk0, kk1, kk2, mscan, present);
    scan3(present, newscan, NN);
    // bf16 prep
    xb_prep_k<<<gN64, B, 0, stream>>>(x, xb);
    w_prep_k<<<3, B, 0, stream>>>(W, Bswz);
    // WMMA GEMM over kept triplets, atomic segment-sum into S
    gemm_k<<<1024, B, 0, stream>>>(xb, (const uint4*)Bswz, kk0, kk1, kk2, invA, mscan, S);
    // finalize outputs
    xnew_k<<<gN64, B, 0, stream>>>(S, cnt, b, segscan, out);
    newei_k<<<gT, B, 0, stream>>>(kk0, kk1, kk2, newscan, mscan, segscan, (int*)d_out);
}
